// GOFuncModel_75076028334661
// MI455X (gfx1250) — matrix operand order repaired and checked by reading, not compiled, not run on hardware
//
#include <hip/hip_runtime.h>
#include <hip/hip_bf16.h>

// ---------------------------------------------------------------------------
// MI455X (gfx1250) RelGraphConv + GO-scoring.
//   stage 0: one-time transpose-convert of weights to bf16 [N,K]
//   stage 1: per-relation edge scatter-add (f32 atomics, HBM-bound)
//   stage 2: 8x bf16 WMMA GEMMs accumulating h [16000,1024] in f32
//   stage 3: bf16 WMMA GEMM vs go_embed (already [N,K]) + sigmoid epilogue
// Hot loop: v_wmma_f32_16x16x32_bf16 fed from LDS, with tiles streamed in by
// GLOBAL_LOAD_ASYNC_TO_LDS_B128 (ASYNCcnt) in a double-buffered pipeline.
// ---------------------------------------------------------------------------

typedef __attribute__((ext_vector_type(16))) __bf16         v16bf;
typedef __attribute__((ext_vector_type(8)))  float          v8f;
typedef __attribute__((ext_vector_type(8)))  unsigned short su8;
typedef __attribute__((ext_vector_type(4)))  unsigned short su4;

union BF16x16 { su8 u[2]; v16bf v; };

__device__ __forceinline__ unsigned short f2bf(float f) {
  // round-to-nearest-even truncation f32 -> bf16
  unsigned u = __float_as_uint(f);
  unsigned r = u + 0x7FFFu + ((u >> 16) & 1u);
  return (unsigned short)(r >> 16);
}

// --------------------------- f32 -> bf16 convert ---------------------------
__global__ __launch_bounds__(256) void k_cvt(const float* __restrict__ in,
                                             unsigned short* __restrict__ out,
                                             int n4) {
  int i = blockIdx.x * 256 + threadIdx.x;
  if (i >= n4) return;
  float4 v = ((const float4*)in)[i];
  su4 o;
  o[0] = f2bf(v.x); o[1] = f2bf(v.y); o[2] = f2bf(v.z); o[3] = f2bf(v.w);
  ((su4*)out)[i] = o;
}

// --------------- f32 [K,N] -> bf16 [N,K] tiled transpose-convert -----------
__global__ __launch_bounds__(256) void k_tcvt(const float* __restrict__ in,
                                              unsigned short* __restrict__ out,
                                              int K, int N) {
  __shared__ unsigned short t[32][33];
  const int k0 = blockIdx.y * 32, n0 = blockIdx.x * 32;
  const int r  = threadIdx.x >> 3;         // 0..31
  const int c4 = (threadIdx.x & 7) << 2;   // 0,4,...,28
  float4 v = *(const float4*)(in + (size_t)(k0 + r) * N + n0 + c4);
  t[r][c4 + 0] = f2bf(v.x);
  t[r][c4 + 1] = f2bf(v.y);
  t[r][c4 + 2] = f2bf(v.z);
  t[r][c4 + 3] = f2bf(v.w);
  __syncthreads();
  su4 o;
  o[0] = t[c4 + 0][r]; o[1] = t[c4 + 1][r]; o[2] = t[c4 + 2][r]; o[3] = t[c4 + 3][r];
  *(su4*)(out + (size_t)(n0 + r) * K + k0 + c4) = o;
}

// ------------------------------- zero f32 ----------------------------------
__global__ __launch_bounds__(256) void k_zero(float4* __restrict__ p, int n4) {
  int i = blockIdx.x * 256 + threadIdx.x;
  if (i < n4) p[i] = make_float4(0.f, 0.f, 0.f, 0.f);
}

// ------------------ edge scatter: agg[dst] += feat[src] --------------------
__global__ __launch_bounds__(256) void k_scatter(const float* __restrict__ feat,
                                                 const int* __restrict__ src,
                                                 const int* __restrict__ dst,
                                                 const int* __restrict__ ety,
                                                 int rel,
                                                 float* __restrict__ agg) {
  const int e = blockIdx.x;
  if (ety[e] != rel) return;
  const int s = src[e], d = dst[e];
  const float4 v = ((const float4*)(feat + (size_t)s * 1024))[threadIdx.x];
  float* p = agg + (size_t)d * 1024 + threadIdx.x * 4;
  atomicAdd(p + 0, v.x);
  atomicAdd(p + 1, v.y);
  atomicAdd(p + 2, v.z);
  atomicAdd(p + 3, v.w);
}

// --------------------------- bf16 WMMA GEMM --------------------------------
// C[M,N] (f32) = A[M,K] (bf16 row-major) x B[N,K] (bf16 row-major, i.e. B^T)
// EPI: 0 = none, 1 = +bias[col], 2 = sigmoid(x + |V[col]|);  ACC: C += ...
// Block 256 thr = 8 waves; tile 128x128, K-step 32; wave tile 64x32 (4x2 WMMA)
// Tiles streamed global->LDS with GLOBAL_LOAD_ASYNC_TO_LDS_B128, double-buffer.
template <int EPI, bool ACC>
__global__ __launch_bounds__(256) void k_gemm(const unsigned short* __restrict__ A,
                                              const unsigned short* __restrict__ B,
                                              float* __restrict__ C,
                                              const float* __restrict__ V,
                                              int M, int N, int K) {
  constexpr int LDS_S = 40;  // 80B row stride: 16B-aligned, conflict-free
  __shared__ __align__(16) unsigned short As[2][128 * LDS_S];
  __shared__ __align__(16) unsigned short Bs[2][128 * LDS_S];

  const int tid  = threadIdx.x;
  const int lane = tid & 31;
  const int wave = tid >> 5;
  const int wm   = wave >> 2;   // 0..1
  const int wn   = wave & 3;    // 0..3
  const int l16  = lane & 15;
  const int lh   = lane >> 4;   // K-half select
  const int bm   = blockIdx.y * 128;
  const int bn   = blockIdx.x * 128;

  // per-thread tile-feed addresses: thread covers 32B of one row (A) + one row (B)
  const int rowf = tid >> 1;           // 0..127
  const int kc   = (tid & 1) << 4;     // 0 or 16 (bf16 elements)
  const unsigned long long gA = (unsigned long long)(A + (size_t)(bm + rowf) * K + kc);
  const unsigned long long gB = (unsigned long long)(B + (size_t)(bn + rowf) * K + kc);
  const unsigned ldsA[2] = {(unsigned)(size_t)&As[0][rowf * LDS_S + kc],
                            (unsigned)(size_t)&As[1][rowf * LDS_S + kc]};
  const unsigned ldsB[2] = {(unsigned)(size_t)&Bs[0][rowf * LDS_S + kc],
                            (unsigned)(size_t)&Bs[1][rowf * LDS_S + kc]};

  // async global->LDS: INST_OFFSET is added to BOTH lds and memory address,
  // so each thread issues its 32 bytes as offset:0 + offset:16.
  auto issue = [&](int kt, int p) {
    unsigned long long a0 = gA + (unsigned long long)kt * 64ull;  // 32 bf16 = 64B
    unsigned long long b0 = gB + (unsigned long long)kt * 64ull;
    asm volatile("global_load_async_to_lds_b128 %0, %1, off"
                 :: "v"(ldsA[p]), "v"(a0) : "memory");
    asm volatile("global_load_async_to_lds_b128 %0, %1, off offset:16"
                 :: "v"(ldsA[p]), "v"(a0) : "memory");
    asm volatile("global_load_async_to_lds_b128 %0, %1, off"
                 :: "v"(ldsB[p]), "v"(b0) : "memory");
    asm volatile("global_load_async_to_lds_b128 %0, %1, off offset:16"
                 :: "v"(ldsB[p]), "v"(b0) : "memory");
  };

  v8f acc[4][2];
#pragma unroll
  for (int i = 0; i < 4; ++i)
#pragma unroll
    for (int j = 0; j < 2; ++j) acc[i][j] = v8f{};

  const int NT = K >> 5;
  issue(0, 0);
  for (int kt = 0; kt < NT; ++kt) {
    const int p = kt & 1;
    // tile kt was issued one iteration ago; finish our lanes, then publish.
    asm volatile("s_wait_asynccnt 0x0" ::: "memory");
    __syncthreads();
    // prefetch tile kt+1 into the other buffer (its last readers finished
    // before the barrier above), hidden behind the WMMA work below.
    if (kt + 1 < NT) issue(kt + 1, p ^ 1);

    // CDNA5 16-bit A layout: lane l -> row l%16; K = {8*lh..+7, 16+8*lh..+7}
    BF16x16 a[4], b[2];
#pragma unroll
    for (int i = 0; i < 4; ++i) {
      const int row = wm * 64 + i * 16 + l16;
      a[i].u[0] = *(const su8*)&As[p][row * LDS_S + lh * 8];
      a[i].u[1] = *(const su8*)&As[p][row * LDS_S + 16 + lh * 8];
    }
    // CDNA5 16-bit B layout: lane l -> col l%16; K = {8*lh..+7, 16+8*lh..+7}
#pragma unroll
    for (int j = 0; j < 2; ++j) {
      const int col = wn * 32 + j * 16 + l16;
      b[j].u[0] = *(const su8*)&Bs[p][col * LDS_S + lh * 8];
      b[j].u[1] = *(const su8*)&Bs[p][col * LDS_S + 16 + lh * 8];
    }
#pragma unroll
    for (int i = 0; i < 4; ++i)
#pragma unroll
      for (int j = 0; j < 2; ++j)
        acc[i][j] = __builtin_amdgcn_wmma_f32_16x16x32_bf16(
            false, a[i].v, false, b[j].v, (short)0, acc[i][j], false, false);
  }

  // C layout: VGPR v -> row v + 8*lh, col = lane%16
#pragma unroll
  for (int i = 0; i < 4; ++i) {
#pragma unroll
    for (int j = 0; j < 2; ++j) {
      const int col = bn + wn * 32 + j * 16 + l16;
      float vb = 0.f;
      if constexpr (EPI == 1) vb = V[col];
      if constexpr (EPI == 2) vb = fabsf(V[col]);
#pragma unroll
      for (int r = 0; r < 8; ++r) {
        const int row = bm + wm * 64 + i * 16 + lh * 8 + r;
        const size_t idx = (size_t)row * N + col;
        float x = acc[i][j][r];
        if constexpr (ACC) x += C[idx];
        if constexpr (EPI == 1) x += vb;
        if constexpr (EPI == 2) x = 1.0f / (1.0f + __expf(-(x + vb)));
        C[idx] = x;
      }
    }
  }
}

// ---------------------------------------------------------------------------
extern "C" void kernel_launch(void* const* d_in, const int* in_sizes, int n_in,
                              void* d_out, int out_size, void* d_ws, size_t ws_size,
                              hipStream_t stream) {
  const float* feat   = (const float*)d_in[0];
  const int*   src    = (const int*)d_in[1];
  const int*   dst    = (const int*)d_in[2];
  const int*   ety    = (const int*)d_in[3];
  const float* W_rel  = (const float*)d_in[4];
  const float* W_loop = (const float*)d_in[5];
  const float* bias   = (const float*)d_in[6];
  const float* go_emb = (const float*)d_in[7];
  const float* go_rad = (const float*)d_in[8];
  float* out = (float*)d_out;

  constexpr int NN = 16000, IF = 1024, OF = 1024, NR = 7, NG = 4096;
  const int E = in_sizes[1];

  // workspace carve-out (~255 MB)
  char* w = (char*)d_ws;
  auto alloc = [&](size_t bytes) {
    char* p = w;
    w += (bytes + 255) & ~(size_t)255;
    return p;
  };
  unsigned short* featb  = (unsigned short*)alloc((size_t)NN * IF * 2);
  unsigned short* WrelT  = (unsigned short*)alloc((size_t)NR * IF * OF * 2);  // [r][OF][IF]
  unsigned short* WloopT = (unsigned short*)alloc((size_t)IF * OF * 2);       // [OF][IF]
  unsigned short* gob    = (unsigned short*)alloc((size_t)NG * OF * 2);       // [NG][OF]
  unsigned short* aggb   = (unsigned short*)alloc((size_t)NN * IF * 2);
  unsigned short* hb     = (unsigned short*)alloc((size_t)NN * OF * 2);
  float*          agg    = (float*)alloc((size_t)NN * IF * 4);
  float*          h      = (float*)alloc((size_t)NN * OF * 4);

  auto cvt = [&](const float* in, unsigned short* outp, size_t n) {
    int n4 = (int)(n / 4);
    k_cvt<<<(n4 + 255) / 256, 256, 0, stream>>>(in, outp, n4);
  };

  // bf16 conversions; weights transposed to [N,K] so all GEMM B-loads are b128
  cvt(feat,   featb, (size_t)NN * IF);
  cvt(go_emb, gob,   (size_t)NG * OF);  // go_embed already [G,K]
  k_tcvt<<<dim3(OF / 32, IF / 32), 256, 0, stream>>>(W_loop, WloopT, IF, OF);
  for (int r = 0; r < NR; ++r)
    k_tcvt<<<dim3(OF / 32, IF / 32), 256, 0, stream>>>(
        W_rel + (size_t)r * IF * OF, WrelT + (size_t)r * OF * IF, IF, OF);

  // h = feat @ W_loop + bias
  k_gemm<1, false><<<dim3(OF / 128, NN / 128), 256, 0, stream>>>(
      featb, WloopT, h, bias, NN, OF, IF);

  // h += segment_sum_r(feat[src]) @ W_rel[r]  for each relation
  for (int r = 0; r < NR; ++r) {
    int n4 = NN * IF / 4;
    k_zero<<<(n4 + 255) / 256, 256, 0, stream>>>((float4*)agg, n4);
    k_scatter<<<E, 256, 0, stream>>>(feat, src, dst, ety, r, agg);
    cvt(agg, aggb, (size_t)NN * IF);
    k_gemm<0, true><<<dim3(OF / 128, NN / 128), 256, 0, stream>>>(
        aggb, WrelT + (size_t)r * OF * IF, h, nullptr, NN, OF, IF);
  }

  // logits = sigmoid(h @ go_embed^T + |rad|)
  cvt(h, hb, (size_t)NN * OF);
  k_gemm<2, false><<<dim3(NG / 128, NN / 128), 256, 0, stream>>>(
      hb, gob, out, go_rad, NN, NG, OF);
}
